// SpMiniUNet_6416681140940
// MI455X (gfx1250) — matrix operand order repaired and checked by reading, not compiled, not run on hardware
//
#include <hip/hip_runtime.h>

typedef __attribute__((ext_vector_type(16))) _Float16 v16h;
typedef __attribute__((ext_vector_type(8)))  _Float16 v8h;
typedef __attribute__((ext_vector_type(8)))  float    v8f;

#define DF 32
#define HF 256
#define WF 320
#define NF (DF*HF*WF)
#define DH2 16
#define HH2 128
#define WH2 160
#define NH (DH2*HH2*WH2)
#define BN_EPS 1e-5f

#define SM_DATA  12096          // 9*84*16 halfs: staged tile
#define SM_ZERO  1344           // zero region (covers SM_DATA + colm<<lcin + 8)

// half-offset of a tap's voxel (relative to the lane's column offset colm<<lcin)
__device__ constexpr int tap_base_c(int tap, int lcin) {
  return ((((tap / 9) * 3 + (tap % 9) / 3) * 84 + (tap % 9) % 3 - 1) << lcin);
}
// padding taps -> zero region
__device__ constexpr int tb_or_pad(int tap, int lcin) {
  return tap < 27 ? tap_base_c(tap, lcin) : SM_DATA;
}

// ---------------------------------------------------------------------------
// Pack conv weights (27 taps, Cin, Cout) -> WMMA B-fragment order, f16.
// B layout (16-bit, 32x16): lane l holds column N = l&15; half j holds
// K = 16*(j>>3) + 8*(l>>4) + (j&7).  Kg = chunk*32 + K; tap = Kg>>lcin, ci = Kg&(Cin-1).
// ---------------------------------------------------------------------------
__global__ void pack_weights_k(const float* __restrict__ w, _Float16* __restrict__ wf,
                               int Cin, int Cout, int nchunks, int lcin, int ntaps) {
  int idx = blockIdx.x * blockDim.x + threadIdx.x;
  if (idx >= nchunks * 512) return;
  int j     = idx & 15;
  int lane  = (idx >> 4) & 31;
  int chunk = idx >> 9;
  int N  = lane & 15;
  int K  = ((j >> 3) << 4) + ((lane >> 4) << 3) + (j & 7);
  int Kg = chunk * 32 + K;
  int tap = Kg >> lcin;
  int ci  = Kg & (Cin - 1);
  float v = 0.f;
  if (tap < ntaps && N < Cout) v = w[(tap * Cin + ci) * Cout + N];
  wf[idx] = (_Float16)v;
}

__global__ void f32_to_f16_k(const float* __restrict__ x, _Float16* __restrict__ y, int n) {
  int i = blockIdx.x * blockDim.x + threadIdx.x;
  if (i < n) y[i] = (_Float16)x[i];
}

// ---------------------------------------------------------------------------
// 3x3x3 submanifold conv + BN + ReLU + mask, implicit GEMM on WMMA f16.
// Block = 160 threads (5 waves); wave -> 16 W-voxels x 16 couts.
// Templated on LCIN/COUT; chunk loop fully unrolled so every tap base is an
// immediate.  For Cin>=8 each A fragment is exactly two 16B-aligned
// ds_load_b128 from the staged tile (padding taps hit the LDS zero region).
// ---------------------------------------------------------------------------
template<int LCIN, int COUT>
__global__ __launch_bounds__(160)
void conv3_wmma_k(const _Float16* __restrict__ in, const _Float16* __restrict__ wfrag,
                  const float* __restrict__ gam, const float* __restrict__ bet,
                  const float* __restrict__ mu,  const float* __restrict__ var,
                  const unsigned char* __restrict__ mask, _Float16* __restrict__ out,
                  int Dd, int Hd, int Wd) {
  constexpr int CIN   = 1 << LCIN;
  constexpr int NCH   = (27 * CIN + 31) >> 5;
  constexpr int LCOUT = (COUT == 16) ? 4 : 3;
  __shared__ _Float16 smem[SM_DATA + SM_ZERO];
  const int tid  = threadIdx.x;
  const int d    = blockIdx.z;
  const int h    = blockIdx.y;
  const int wblk = blockIdx.x * 80;

  // ---- zero region ----
  for (int i = tid; i < SM_ZERO / 2; i += 160)
    *(unsigned int*)(smem + SM_DATA + 2 * i) = 0u;

  // ---- stage input tile (f16, zero-filled halo) ----
  constexpr int CW  = CIN >> 1;               // u32 units per voxel
  constexpr int NST = 9 * 82 * CW;
  for (int i = tid; i < NST; i += 160) {
    int c2 = i % CW;
    int t  = i / CW;
    int wc = t % 82;
    int zy = t / 82;
    int z = d + zy / 3 - 1;
    int y = h + zy % 3 - 1;
    int w = wblk + wc - 1;
    unsigned int val = 0u;
    if (z >= 0 && z < Dd && y >= 0 && y < Hd && w >= 0 && w < Wd)
      val = *(const unsigned int*)(in + ((((long long)z * Hd + y) * Wd + w) << LCIN) + (c2 << 1));
    *(unsigned int*)(smem + (((zy * 84 + wc) << LCIN) + (c2 << 1))) = val;
  }
  __syncthreads();

  const int lane = tid & 31;
  const int wave = tid >> 5;
  const int m    = lane & 15;
  const int g2   = lane >> 4;
  const int colm = wave * 16 + 1 + m;
  // per-lane half offset; for CIN==16 the ci base 8*g2 folds in here
  const int cofs = (colm << LCIN) + ((LCIN == 4) ? (g2 << 3) : 0);

  v8f acc = {0.f, 0.f, 0.f, 0.f, 0.f, 0.f, 0.f, 0.f};
  const _Float16* wp = wfrag + (lane << 4);
#pragma unroll
  for (int kc = 0; kc < NCH; ++kc) {
    v16h b = *(const v16h*)(wp + (kc << 9));        // B fragment (L2-resident)
    v16h a;
    if (LCIN == 4) {
      // halfs 0-7: tap 2kc, ci 8g2..; halfs 8-15: tap 2kc+1
      *((v8h*)&a)       = *(const v8h*)(smem + tb_or_pad(2 * kc,     4) + cofs);
      *(((v8h*)&a) + 1) = *(const v8h*)(smem + tb_or_pad(2 * kc + 1, 4) + cofs);
    } else if (LCIN == 3) {
      // halfs 0-7: tap 4kc+g2, ci 0..7; halfs 8-15: tap 4kc+2+g2
      int ba = g2 ? tb_or_pad(4 * kc + 1, 3) : tb_or_pad(4 * kc,     3);
      int bb = g2 ? tb_or_pad(4 * kc + 3, 3) : tb_or_pad(4 * kc + 2, 3);
      *((v8h*)&a)       = *(const v8h*)(smem + ba + cofs);
      *(((v8h*)&a) + 1) = *(const v8h*)(smem + bb + cofs);
    } else {
      // CIN==2: pairwise b32 gathers; all bases compile-time constants
      unsigned int* ap = (unsigned int*)&a;
#pragma unroll
      for (int p = 0; p < 8; ++p) {
        int jj  = p << 1;
        int K00 = ((jj >> 3) << 4) + (jj & 7);      // K without the 8*g2 term
        int Kg0 = kc * 32 + K00;
        int b0v = tb_or_pad((Kg0)     >> LCIN, LCIN) + ((Kg0)     & (CIN - 1));
        int b1v = tb_or_pad((Kg0 + 8) >> LCIN, LCIN) + ((Kg0 + 8) & (CIN - 1));
        int bse = g2 ? b1v : b0v;
        ap[p] = *(const unsigned int*)(smem + bse + cofs);
      }
    }
    acc = __builtin_amdgcn_wmma_f32_16x16x32_f16(false, a, false, b,
                                                 (short)0, acc, false, false);
  }

  // ---- fused BN + ReLU + mask, store f16 ----
  const int N = lane & 15;
  float sc = 0.f, sh = 0.f;
  if (N < COUT) {
    sc = gam[N] * rsqrtf(var[N] + BN_EPS);
    sh = bet[N] - mu[N] * sc;
  }
  long long vb = ((long long)d * Hd + h) * Wd + wblk + wave * 16 + g2 * 8;
  uint2 mv = *(const uint2*)(mask + vb);            // 8 mask bytes, 8B aligned
  unsigned long long mbits = ((unsigned long long)mv.y << 32) | (unsigned long long)mv.x;
#pragma unroll
  for (int r = 0; r < 8; ++r) {
    if (N < COUT) {
      bool mk = (mbits >> (8 * r)) & 0xFFull;
      float yv = fmaxf(acc[r] * sc + sh, 0.f);
      out[((vb + r) << LCOUT) + N] = (_Float16)(mk ? yv : 0.f);
    }
  }
}

// ---------------------------------------------------------------------------
// down1: 2x2x2 stride-2 conv 8->16, derives half-res mask m2 (OR pooling).
// ---------------------------------------------------------------------------
__global__ void down_k(const _Float16* __restrict__ in, const float* __restrict__ w,
                       const unsigned char* __restrict__ mask,
                       _Float16* __restrict__ out, unsigned char* __restrict__ m2) {
  int idx = blockIdx.x * blockDim.x + threadIdx.x;
  if (idx >= NH * 16) return;
  int co = idx & 15;
  int v  = idx >> 4;
  int wq = v % WH2;
  int t  = v / WH2;
  int hq = t % HH2;
  int dq = t / HH2;
  float acc = 0.f;
  unsigned char mm = 0;
#pragma unroll
  for (int tap = 0; tap < 8; ++tap) {
    int dz = tap >> 2, dy = (tap >> 1) & 1, dx = tap & 1;
    long long mi = (((long long)(2 * dq + dz) * HF) + (2 * hq + dy)) * WF + (2 * wq + dx);
    mm |= mask[mi];
#pragma unroll
    for (int ci = 0; ci < 8; ++ci)
      acc += (float)in[mi * 8 + ci] * w[(tap * 8 + ci) * 16 + co];
  }
  out[(long long)v * 16 + co] = (_Float16)(mm ? acc : 0.f);
  if (co == 0) m2[v] = mm ? 1 : 0;
}

// ---------------------------------------------------------------------------
// up1 (conv_transpose 2x2x2 s2, 16->8) + mask + concat with skip1 -> 16ch.
// Non-overlapping: out[o] uses tap k = 1-(o&1) per dim, input o>>1.
// ---------------------------------------------------------------------------
__global__ void up_concat_k(const _Float16* __restrict__ enc2, const float* __restrict__ w,
                            const _Float16* __restrict__ skip1,
                            const unsigned char* __restrict__ mask,
                            _Float16* __restrict__ cat) {
  long long idx = (long long)blockIdx.x * blockDim.x + threadIdx.x;
  if (idx >= (long long)NF * 8) return;
  int co = (int)(idx & 7);
  long long v = idx >> 3;
  int wq = (int)(v % WF);
  long long t = v / WF;
  int hq = (int)(t % HF);
  int dq = (int)(t / HF);
  long long iv = (((long long)(dq >> 1) * HH2) + (hq >> 1)) * WH2 + (wq >> 1);
  int k = ((1 - (dq & 1)) * 2 + (1 - (hq & 1))) * 2 + (1 - (wq & 1));
  float acc = 0.f;
#pragma unroll
  for (int ci = 0; ci < 16; ++ci)
    acc += (float)enc2[iv * 16 + ci] * w[(k * 16 + ci) * 8 + co];
  unsigned char mm = mask[v];
  cat[v * 16 + co]     = (_Float16)(mm ? acc : 0.f);
  cat[v * 16 + 8 + co] = skip1[v * 8 + co];   // skip1 already masked
}

// ---------------------------------------------------------------------------
// out: 1x1x1 conv 8->8 + bias, masked, f32 output.
// ---------------------------------------------------------------------------
__global__ void out_k(const _Float16* __restrict__ in, const float* __restrict__ w,
                      const float* __restrict__ b, const unsigned char* __restrict__ mask,
                      float* __restrict__ out) {
  long long idx = (long long)blockIdx.x * blockDim.x + threadIdx.x;
  if (idx >= (long long)NF * 8) return;
  int co = (int)(idx & 7);
  long long v = idx >> 3;
  float acc = b[co];
#pragma unroll
  for (int ci = 0; ci < 8; ++ci)
    acc += (float)in[v * 8 + ci] * w[ci * 8 + co];
  out[idx] = mask[v] ? acc : 0.f;
}

// ---------------------------------------------------------------------------
extern "C" void kernel_launch(void* const* d_in, const int* in_sizes, int n_in,
                              void* d_out, int out_size, void* d_ws, size_t ws_size,
                              hipStream_t stream) {
  (void)in_sizes; (void)n_in; (void)out_size; (void)ws_size;
  const float*         x    = (const float*)d_in[0];
  const unsigned char* mask = (const unsigned char*)d_in[1];
  const float* w_e11 = (const float*)d_in[2];
  const float* g11 = (const float*)d_in[3],  *b11 = (const float*)d_in[4];
  const float* m11 = (const float*)d_in[5],  *v11 = (const float*)d_in[6];
  const float* w_e12 = (const float*)d_in[7];
  const float* g12 = (const float*)d_in[8],  *b12 = (const float*)d_in[9];
  const float* m12 = (const float*)d_in[10], *v12 = (const float*)d_in[11];
  const float* w_dn  = (const float*)d_in[12];
  const float* w_e21 = (const float*)d_in[13];
  const float* g21 = (const float*)d_in[14], *b21 = (const float*)d_in[15];
  const float* m21 = (const float*)d_in[16], *v21 = (const float*)d_in[17];
  const float* w_e22 = (const float*)d_in[18];
  const float* g22 = (const float*)d_in[19], *b22 = (const float*)d_in[20];
  const float* m22 = (const float*)d_in[21], *v22 = (const float*)d_in[22];
  const float* w_up  = (const float*)d_in[23];
  const float* w_d11 = (const float*)d_in[24];
  const float* gd1 = (const float*)d_in[25], *bd1 = (const float*)d_in[26];
  const float* md1 = (const float*)d_in[27], *vd1 = (const float*)d_in[28];
  const float* w_d12 = (const float*)d_in[29];
  const float* gd2 = (const float*)d_in[30], *bd2 = (const float*)d_in[31];
  const float* md2 = (const float*)d_in[32], *vd2 = (const float*)d_in[33];
  const float* w_out = (const float*)d_in[34];
  const float* b_out = (const float*)d_in[35];

  // ---- workspace layout ----
  char* ws = (char*)d_ws;
  size_t off = 0;
  auto alloc = [&](size_t bytes) -> char* {
    char* p = ws + off;
    off = (off + bytes + 255) & ~(size_t)255;
    return p;
  };
  _Float16* xh    = (_Float16*)alloc((size_t)NF * 2 * 2);
  _Float16* h1    = (_Float16*)alloc((size_t)NF * 8 * 2);   // enc1 out; reused as dec1 out
  _Float16* skip1 = (_Float16*)alloc((size_t)NF * 8 * 2);
  _Float16* halfA = (_Float16*)alloc((size_t)NH * 16 * 2);
  _Float16* halfB = (_Float16*)alloc((size_t)NH * 16 * 2);
  _Float16* cat   = (_Float16*)alloc((size_t)NF * 16 * 2);  // reused as dec2 out (42MB < 84MB)
  unsigned char* m2 = (unsigned char*)alloc((size_t)NH);
  _Float16* wf_e11 = (_Float16*)alloc(2  * 1024);
  _Float16* wf_e12 = (_Float16*)alloc(7  * 1024);
  _Float16* wf_e21 = (_Float16*)alloc(14 * 1024);
  _Float16* wf_e22 = (_Float16*)alloc(14 * 1024);
  _Float16* wf_d11 = (_Float16*)alloc(14 * 1024);
  _Float16* wf_d12 = (_Float16*)alloc(7  * 1024);
  _Float16* decA = h1;
  _Float16* decB = cat;

  // ---- pack weights + convert input ----
  auto pack = [&](const float* w, _Float16* wf, int Cin, int Cout, int nch, int lcin) {
    int tot = nch * 512;
    pack_weights_k<<<(tot + 255) / 256, 256, 0, stream>>>(w, wf, Cin, Cout, nch, lcin, 27);
  };
  pack(w_e11, wf_e11, 2,  8,  2,  1);
  pack(w_e12, wf_e12, 8,  8,  7,  3);
  pack(w_e21, wf_e21, 16, 16, 14, 4);
  pack(w_e22, wf_e22, 16, 16, 14, 4);
  pack(w_d11, wf_d11, 16, 8,  14, 4);
  pack(w_d12, wf_d12, 8,  8,  7,  3);
  f32_to_f16_k<<<(NF * 2 + 255) / 256, 256, 0, stream>>>(x, xh, NF * 2);

  dim3 gfull(WF / 80, HF, DF);
  dim3 ghalf(WH2 / 80, HH2, DH2);

  // enc1
  conv3_wmma_k<1, 8><<<gfull, 160, 0, stream>>>(xh, wf_e11, g11, b11, m11, v11, mask, h1,    DF, HF, WF);
  conv3_wmma_k<3, 8><<<gfull, 160, 0, stream>>>(h1, wf_e12, g12, b12, m12, v12, mask, skip1, DF, HF, WF);
  // down1 (+ m2)
  down_k<<<(NH * 16 + 255) / 256, 256, 0, stream>>>(skip1, w_dn, mask, halfA, m2);
  // enc2
  conv3_wmma_k<4, 16><<<ghalf, 160, 0, stream>>>(halfA, wf_e21, g21, b21, m21, v21, m2, halfB, DH2, HH2, WH2);
  conv3_wmma_k<4, 16><<<ghalf, 160, 0, stream>>>(halfB, wf_e22, g22, b22, m22, v22, m2, halfA, DH2, HH2, WH2);
  // up1 + concat skip
  up_concat_k<<<(int)(((long long)NF * 8 + 255) / 256), 256, 0, stream>>>(halfA, w_up, skip1, mask, cat);
  // dec1
  conv3_wmma_k<4, 8><<<gfull, 160, 0, stream>>>(cat,  wf_d11, gd1, bd1, md1, vd1, mask, decA, DF, HF, WF);
  conv3_wmma_k<3, 8><<<gfull, 160, 0, stream>>>(decA, wf_d12, gd2, bd2, md2, vd2, mask, decB, DF, HF, WF);
  // out head
  out_k<<<(int)(((long long)NF * 8 + 255) / 256), 256, 0, stream>>>(decB, w_out, b_out, mask, (float*)d_out);
}
